// MultiHeadAttention_72421738545471
// MI455X (gfx1250) — matrix operand order repaired and checked by reading, not compiled, not run on hardware
//
#include <hip/hip_runtime.h>
#include <hip/hip_bf16.h>

// MI455X / gfx1250 fused kernel set for MHA + KNN-memory block.
// All GEMM-shaped math runs on v_wmma_f32_16x16x32_f16 (wave32, 16x16 tiles,
// fp32 accumulate). Softmax / top-k / normalization in fp32 VALU.

typedef __attribute__((ext_vector_type(16))) _Float16 v16h;
typedef __attribute__((ext_vector_type(8)))  _Float16 v8h;
typedef __attribute__((ext_vector_type(8)))  float    v8f;

#define BQ 2
#define SQ 1024
#define DQ 1024
#define HQ 16
#define DK 64
#define MMEM 8192
#define NTOK (BQ * SQ)   // 2048

__device__ __forceinline__ v8f wmma16(v16h a, v16h b, v8f c) {
  return __builtin_amdgcn_wmma_f32_16x16x32_f16(
      /*neg_a=*/false, a, /*neg_b=*/false, b,
      /*c_mod=*/(short)0, c, /*reuse_a=*/false, /*reuse_b=*/false);
}

// A fragment: 16(M) x 32(K) f16, row-major source with leading dim lda.
// Layout: lane l (m=l&15, hi=l>>4): slots 0..15 -> K=16*hi+s
// (matches B fragment). Two adjacent 16B loads per lane.
__device__ __forceinline__ v16h load_a_frag(const _Float16* base, int lda, int kbase) {
  int lane = threadIdx.x & 31;
  int m = lane & 15, hi = lane >> 4;
  const _Float16* p = base + (size_t)m * lda + kbase + 16 * hi;
  v8h lo = *(const v8h*)p;
  v8h hh = *(const v8h*)(p + 8);
  v16h r;
#pragma unroll
  for (int i = 0; i < 8; ++i) { r[i] = lo[i]; r[i + 8] = hh[i]; }
  return r;
}

// B fragment: 32(K) x 16(N). Column N holds a row of the (row-major) source
// matrix: lane l (n=l&15, hi=l>>4): slots 0..15 -> K=16*hi+s. One 32B run.
__device__ __forceinline__ v16h load_b_frag(const _Float16* base, int ldb, int kbase) {
  int lane = threadIdx.x & 31;
  int n = lane & 15, hi = lane >> 4;
  return *(const v16h*)(base + (size_t)n * ldb + kbase + 16 * hi);
}

// ---------------- elementwise / normalization helpers ----------------

__global__ void cvt16_kernel(const float* __restrict__ src, _Float16* __restrict__ dst, int n) {
  int i = blockIdx.x * blockDim.x + threadIdx.x;
  if (i < n) dst[i] = (_Float16)src[i];
}

// one wave per 64-wide row: normalized f16 copy of mem_keys
__global__ void keynorm_kernel(const float* __restrict__ src, _Float16* __restrict__ dst, int nrows) {
  int w = blockIdx.x * (blockDim.x >> 5) + (threadIdx.x >> 5);
  if (w >= nrows) return;
  int lane = threadIdx.x & 31;
  const float* r = src + (size_t)w * DK;
  float a = r[lane], b = r[lane + 32];
  float ss = a * a + b * b;
#pragma unroll
  for (int m = 1; m <= 16; m <<= 1) ss += __shfl_xor(ss, m, 32);
  float sc = 1.0f / (sqrtf(ss) + 1e-8f);
  dst[(size_t)w * DK + lane] = (_Float16)(a * sc);
  dst[(size_t)w * DK + lane + 32] = (_Float16)(b * sc);
}

// one wave per (b,h,s) head-row of q: L2-normalize the 64-slice in place layout [B,S,D]
__global__ void qnorm_kernel(const _Float16* __restrict__ q, _Float16* __restrict__ qn) {
  int w = blockIdx.x * (blockDim.x >> 5) + (threadIdx.x >> 5);  // 0..32767
  if (w >= BQ * HQ * SQ) return;
  int lane = threadIdx.x & 31;
  int s = w & (SQ - 1), h = (w >> 10) & (HQ - 1), b = w >> 14;
  size_t off = ((size_t)(b * SQ + s)) * DQ + h * DK;
  float a = (float)q[off + lane], c = (float)q[off + lane + 32];
  float ss = a * a + c * c;
#pragma unroll
  for (int m = 1; m <= 16; m <<= 1) ss += __shfl_xor(ss, m, 32);
  float sc = 1.0f / (sqrtf(ss) + 1e-8f);
  qn[off + lane] = (_Float16)(a * sc);
  qn[off + lane + 32] = (_Float16)(c * sc);
}

__global__ void combine_kernel(const float* __restrict__ mo, const float* __restrict__ ao,
                               const float* __restrict__ gate, _Float16* __restrict__ z, int n) {
  int i = blockIdx.x * blockDim.x + threadIdx.x;
  if (i >= n) return;
  float g = 1.0f / (1.0f + __expf(-gate[0]));
  z[i] = (_Float16)(g * mo[i] + (1.0f - g) * ao[i]);
}

// ---------------- WMMA GEMM: C = A[N,K] * W[Dout,K]^T + bias ----------------
// MODE 0: f16 row-major out.  MODE 1: f16 per-head transposed (Vt[(b,h,d)][s]).
// MODE 2: fp32 row-major out.
template <int MODE>
__global__ void gemm_wmma_kernel(const _Float16* __restrict__ A, const _Float16* __restrict__ W,
                                 const float* __restrict__ bias, void* __restrict__ Cout,
                                 int Nrows, int K, int Dout) {
  int w = blockIdx.x * (blockDim.x >> 5) + (threadIdx.x >> 5);
  int mtiles = Nrows >> 4;
  int mtile = w % mtiles;
  int nt4 = w / mtiles;          // 64-wide N tile
  int mbase = mtile * 16, nbase = nt4 * 64;
  v8f acc[4] = {v8f{}, v8f{}, v8f{}, v8f{}};
  for (int kb = 0; kb < K; kb += 32) {
    v16h a = load_a_frag(A + (size_t)mbase * K, K, kb);
#pragma unroll
    for (int t = 0; t < 4; ++t) {
      v16h b = load_b_frag(W + (size_t)(nbase + t * 16) * K, K, kb);
      acc[t] = wmma16(a, b, acc[t]);
    }
  }
  int lane = threadIdx.x & 31, hi = lane >> 4, nl = lane & 15;
#pragma unroll
  for (int t = 0; t < 4; ++t) {
#pragma unroll
    for (int j = 0; j < 8; ++j) {
      int row = mbase + j + 8 * hi;
      int col = nbase + t * 16 + nl;
      float v = acc[t][j] + bias[col];
      if (MODE == 0) {
        ((_Float16*)Cout)[(size_t)row * Dout + col] = (_Float16)v;
      } else if (MODE == 1) {
        int bb = row >> 10, s = row & (SQ - 1);
        int h = col >> 6, d = col & (DK - 1);
        ((_Float16*)Cout)[(((size_t)((bb * HQ + h) * DK + d)) << 10) + s] = (_Float16)v;
      } else {
        ((float*)Cout)[(size_t)row * Dout + col] = v;
      }
    }
  }
}

// ---------------- flash attention: one wave = (b,h) x 16 queries ----------------
__global__ void attn_kernel(const _Float16* __restrict__ Qh, const _Float16* __restrict__ Kh,
                            const _Float16* __restrict__ Vt, float* __restrict__ Oout) {
  __shared__ __align__(16) _Float16 plds[4][16][32];
  int wInB = threadIdx.x >> 5;
  int w = blockIdx.x * (blockDim.x >> 5) + wInB;
  int qt = w & 63, h = (w >> 6) & (HQ - 1), b = w >> 10;
  int qbase = qt * 16;
  int lane = threadIdx.x & 31, hi = lane >> 4, nl = lane & 15;

  const _Float16* qptr = Qh + ((size_t)(b * SQ + qbase)) * DQ + h * DK;
  v16h aq0 = load_a_frag(qptr, DQ, 0);
  v16h aq1 = load_a_frag(qptr, DQ, 32);

  float mrow[8], lrow[8];
#pragma unroll
  for (int j = 0; j < 8; ++j) { mrow[j] = -3e38f; lrow[j] = 0.0f; }
  v8f o[4] = {v8f{}, v8f{}, v8f{}, v8f{}};

  int last_kt = (qbase + 15) >> 5;
  for (int kt = 0; kt <= last_kt; ++kt) {
    const _Float16* kptr = Kh + ((size_t)(b * SQ + kt * 32)) * DQ + h * DK;
    v16h bk00 = load_b_frag(kptr, DQ, 0);
    v16h bk01 = load_b_frag(kptr, DQ, 32);
    v16h bk10 = load_b_frag(kptr + (size_t)16 * DQ, DQ, 0);
    v16h bk11 = load_b_frag(kptr + (size_t)16 * DQ, DQ, 32);
    v8f s0 = {}, s1 = {};
    s0 = wmma16(aq0, bk00, s0); s0 = wmma16(aq1, bk01, s0);
    s1 = wmma16(aq0, bk10, s1); s1 = wmma16(aq1, bk11, s1);

#pragma unroll
    for (int j = 0; j < 8; ++j) {
      int qrow = qbase + j + 8 * hi;
      float v0 = s0[j] * 0.125f;                       // 1/sqrt(64)
      float v1 = s1[j] * 0.125f;
      if (kt * 32 + nl > qrow)      v0 = -1e9f;        // causal NEG_INF
      if (kt * 32 + 16 + nl > qrow) v1 = -1e9f;
      float vmax = fmaxf(v0, v1);
#pragma unroll
      for (int m = 1; m <= 8; m <<= 1) vmax = fmaxf(vmax, __shfl_xor(vmax, m, 32));
      float mnew = fmaxf(mrow[j], vmax);
      float corr = __expf(mrow[j] - mnew);
      float p0 = __expf(v0 - mnew);
      float p1 = __expf(v1 - mnew);
      float rs = p0 + p1;
#pragma unroll
      for (int m = 1; m <= 8; m <<= 1) rs += __shfl_xor(rs, m, 32);
      lrow[j] = lrow[j] * corr + rs;
      mrow[j] = mnew;
      o[0][j] *= corr; o[1][j] *= corr; o[2][j] *= corr; o[3][j] *= corr;
      plds[wInB][j + 8 * hi][nl] = (_Float16)p0;
      plds[wInB][j + 8 * hi][16 + nl] = (_Float16)p1;
    }
    asm volatile("s_wait_dscnt 0" ::: "memory");
    v16h pa = load_a_frag(&plds[wInB][0][0], 32, 0);   // P in A-frag layout

    const _Float16* vptr = Vt + ((size_t)((b * HQ + h) * DK)) * SQ + kt * 32;
#pragma unroll
    for (int t = 0; t < 4; ++t) {
      v16h bv = load_b_frag(vptr + (size_t)(t * 16) * SQ, SQ, 0);
      o[t] = wmma16(pa, bv, o[t]);
    }
  }
#pragma unroll
  for (int t = 0; t < 4; ++t)
#pragma unroll
    for (int j = 0; j < 8; ++j) {
      int row = qbase + j + 8 * hi;
      Oout[((size_t)(b * SQ + row)) * DQ + h * DK + t * 16 + nl] = o[t][j] / lrow[j];
    }
}

// ---------------- KNN memory: one wave = (b,h) x 32 queries ----------------
__global__ void knn_kernel(const _Float16* __restrict__ Qn, const _Float16* __restrict__ MK,
                           const float* __restrict__ MV, float* __restrict__ Mout) {
  __shared__ float slds[4][32][33];
  int wInB = threadIdx.x >> 5;
  int w = blockIdx.x * (blockDim.x >> 5) + wInB;
  int st = w & 31, h = (w >> 5) & (HQ - 1), b = w >> 9;
  int sbase = st * 32;
  int lane = threadIdx.x & 31, hi = lane >> 4, nl = lane & 15;

  const _Float16* qptr = Qn + ((size_t)(b * SQ + sbase)) * DQ + h * DK;
  v16h a0k0 = load_a_frag(qptr, DQ, 0);
  v16h a0k1 = load_a_frag(qptr, DQ, 32);
  v16h a1k0 = load_a_frag(qptr + (size_t)16 * DQ, DQ, 0);
  v16h a1k1 = load_a_frag(qptr + (size_t)16 * DQ, DQ, 32);

  float ts[32]; int ti[32];
#pragma unroll
  for (int i = 0; i < 32; ++i) { ts[i] = -1e30f; ti[i] = 0; }
  float minv = -1e30f; int minp = 0;

  for (int mt = 0; mt < MMEM / 32; ++mt) {
    const _Float16* kp = MK + (size_t)(mt * 32) * DK;
    v16h b00 = load_b_frag(kp, DK, 0);
    v16h b01 = load_b_frag(kp, DK, 32);
    v16h b10 = load_b_frag(kp + (size_t)16 * DK, DK, 0);
    v16h b11 = load_b_frag(kp + (size_t)16 * DK, DK, 32);
    v8f s00 = {}, s01 = {}, s10 = {}, s11 = {};
    s00 = wmma16(a0k0, b00, s00); s00 = wmma16(a0k1, b01, s00);
    s01 = wmma16(a0k0, b10, s01); s01 = wmma16(a0k1, b11, s01);
    s10 = wmma16(a1k0, b00, s10); s10 = wmma16(a1k1, b01, s10);
    s11 = wmma16(a1k0, b10, s11); s11 = wmma16(a1k1, b11, s11);
#pragma unroll
    for (int j = 0; j < 8; ++j) {
      slds[wInB][j + 8 * hi][nl]           = s00[j];
      slds[wInB][j + 8 * hi][16 + nl]      = s01[j];
      slds[wInB][16 + j + 8 * hi][nl]      = s10[j];
      slds[wInB][16 + j + 8 * hi][16 + nl] = s11[j];
    }
    asm volatile("s_wait_dscnt 0" ::: "memory");
    // lane scans its own query row, maintains private top-32
    for (int c = 0; c < 32; ++c) {
      float sv = slds[wInB][lane][c];
      if (sv > minv) {
        ts[minp] = sv; ti[minp] = mt * 32 + c;
        minv = ts[0]; minp = 0;
        for (int jj = 1; jj < 32; ++jj)
          if (ts[jj] < minv) { minv = ts[jj]; minp = jj; }
      }
    }
    asm volatile("s_wait_dscnt 0" ::: "memory");
  }

  float acc[DK];
#pragma unroll
  for (int d = 0; d < DK; ++d) acc[d] = 0.0f;
  for (int i = 0; i < 32; ++i) {
    float s = ts[i];
    const float* vp = MV + (size_t)ti[i] * DK;
#pragma unroll
    for (int d = 0; d < DK; ++d) acc[d] += s * vp[d];
  }
  float* op = Mout + ((size_t)(b * SQ + sbase + lane)) * DQ + h * DK;
#pragma unroll
  for (int d = 0; d < DK; ++d) op[d] = acc[d];
}

// ---------------- host orchestration ----------------

extern "C" void kernel_launch(void* const* d_in, const int* in_sizes, int n_in,
                              void* d_out, int out_size, void* d_ws, size_t ws_size,
                              hipStream_t stream) {
  const float* x    = (const float*)d_in[0];
  const float* Wq   = (const float*)d_in[1];
  const float* bq   = (const float*)d_in[2];
  const float* Wk   = (const float*)d_in[3];
  const float* bk   = (const float*)d_in[4];
  const float* Wv   = (const float*)d_in[5];
  const float* bv   = (const float*)d_in[6];
  const float* Wo   = (const float*)d_in[7];
  const float* bo   = (const float*)d_in[8];
  const float* mk   = (const float*)d_in[9];
  const float* mv   = (const float*)d_in[10];
  const float* gate = (const float*)d_in[11];

  char* ws = (char*)d_ws;
  const size_t MB = 1u << 20;
  _Float16* xh  = (_Float16*)(ws + 0 * MB);    // 4 MiB  [NTOK x D] f16
  _Float16* wqh = (_Float16*)(ws + 4 * MB);    // 2 MiB
  _Float16* wkh = (_Float16*)(ws + 6 * MB);
  _Float16* wvh = (_Float16*)(ws + 8 * MB);
  _Float16* woh = (_Float16*)(ws + 10 * MB);
  _Float16* qh  = (_Float16*)(ws + 12 * MB);   // 4 MiB
  _Float16* kh  = (_Float16*)(ws + 16 * MB);   // 4 MiB
  _Float16* vth = (_Float16*)(ws + 20 * MB);   // 4 MiB  Vt[(b,h,d)][s]
  _Float16* qnh = (_Float16*)(ws + 24 * MB);   // 4 MiB
  _Float16* mkh = (_Float16*)(ws + 28 * MB);   // 1 MiB  normalized mem_keys f16
  float*    ao  = (float*)(ws + 32 * MB);      // 8 MiB  attn_out (pre-Wo)
  float*    mo  = (float*)(ws + 40 * MB);      // 8 MiB  memory_out (pre-Wo)
  _Float16* zh  = xh;                          // reuse x_f16 region (dead by then)

  const int NEL = NTOK * DQ;     // 2097152
  const int WEL = DQ * DQ;       // 1048576

  cvt16_kernel<<<(NEL + 255) / 256, 256, 0, stream>>>(x, xh, NEL);
  cvt16_kernel<<<(WEL + 255) / 256, 256, 0, stream>>>(Wq, wqh, WEL);
  cvt16_kernel<<<(WEL + 255) / 256, 256, 0, stream>>>(Wk, wkh, WEL);
  cvt16_kernel<<<(WEL + 255) / 256, 256, 0, stream>>>(Wv, wvh, WEL);
  cvt16_kernel<<<(WEL + 255) / 256, 256, 0, stream>>>(Wo, woh, WEL);
  keynorm_kernel<<<MMEM / 8, 256, 0, stream>>>(mk, mkh, MMEM);

  // projections: 2048 waves each = 512 blocks x 4 waves
  gemm_wmma_kernel<0><<<512, 128, 0, stream>>>(xh, wqh, bq, qh,  NTOK, DQ, DQ);
  gemm_wmma_kernel<0><<<512, 128, 0, stream>>>(xh, wkh, bk, kh,  NTOK, DQ, DQ);
  gemm_wmma_kernel<1><<<512, 128, 0, stream>>>(xh, wvh, bv, vth, NTOK, DQ, DQ);

  qnorm_kernel<<<4096, 256, 0, stream>>>(qh, qnh);           // 32768 waves
  attn_kernel<<<512, 128, 0, stream>>>(qh, kh, vth, ao);     // 2048 waves
  knn_kernel<<<256, 128, 0, stream>>>(qnh, mkh, mv, mo);     // 1024 waves

  combine_kernel<<<(NEL + 255) / 256, 256, 0, stream>>>(mo, ao, gate, zh, NEL);
  gemm_wmma_kernel<2><<<512, 128, 0, stream>>>(zh, woh, bo, (float*)d_out, NTOK, DQ, DQ);
}
